// NonLocal_51075751084594
// MI455X (gfx1250) — compile-verified, matrix operand order checked by reading, not run on hardware
//
#include <hip/hip_runtime.h>
#include <hip/hip_bf16.h>
#include <stdint.h>

typedef __attribute__((ext_vector_type(16))) __bf16 v16bf;
typedef __attribute__((ext_vector_type(8)))  float  v8f;

#define HW    4096
#define NB    8
#define CIN   256
#define CD    128   // C2
#define QTILE 128
#define KTILE 64

union BFrag {
  v16bf v;
  unsigned int u[8];
};

// Build a 16-element (8-dword) bf16 fragment for one row, following the
// CDNA5 16-bit A/B layout: lane-half h reads K = {base + h*8 + 0..7} and
// K = {base + 16 + h*8 + 0..7}, packed two per dword.
__device__ __forceinline__ void load_frag_row(BFrag& f, const unsigned int* row_dw,
                                              int coff_dw, int half) {
#pragma unroll
  for (int i = 0; i < 4; ++i) f.u[i]     = row_dw[coff_dw + half * 4 + i];
#pragma unroll
  for (int i = 0; i < 4; ++i) f.u[4 + i] = row_dw[coff_dw + 8 + half * 4 + i];
}

// ---------------------------------------------------------------------------
// Kernel A: 1x1 conv projection.  out = w[CD,CIN] @ x[b,CIN,HW] + bias
// qmajor=1 -> out[b][q][c] (for Q,K); qmajor=0 -> out[b][c][q] (for V)
// ---------------------------------------------------------------------------
__global__ void __launch_bounds__(256) proj_kernel(
    const float* __restrict__ x, const float* __restrict__ w,
    const float* __restrict__ bias, __bf16* __restrict__ out, int qmajor)
{
  __shared__ float xs[64][65];
  const int t  = threadIdx.x;
  const int b  = blockIdx.x >> 6;          // 64 q-tiles per batch
  const int q0 = (blockIdx.x & 63) << 6;   // 64 spatial positions per block
  const int c     = t >> 1;                // 0..127 output channel
  const int qbase = (t & 1) << 5;          // 0 or 32

  float acc[32];
#pragma unroll
  for (int j = 0; j < 32; ++j) acc[j] = 0.f;

  for (int chunk = 0; chunk < 4; ++chunk) {
    const int cin0 = chunk << 6;
#pragma unroll
    for (int i = 0; i < 16; ++i) {
      int idx = t + (i << 8);
      int ci = idx >> 6, q = idx & 63;
      xs[ci][q] = x[((size_t)(b * CIN + cin0 + ci)) * HW + q0 + q];
    }
    __syncthreads();
    for (int ci = 0; ci < 64; ++ci) {
      float wv = w[c * CIN + cin0 + ci];
#pragma unroll
      for (int j = 0; j < 32; ++j) acc[j] += wv * xs[ci][qbase + j];
    }
    __syncthreads();
  }

  const float bb = bias[c];
  if (qmajor) {
#pragma unroll
    for (int j = 0; j < 32; ++j) {
      int q = q0 + qbase + j;
      out[((size_t)(b * HW) + q) * CD + c] = (__bf16)(acc[j] + bb);
    }
  } else {
#pragma unroll
    for (int j = 0; j < 32; ++j)
      out[((size_t)(b * CD + c)) * HW + q0 + qbase + j] = (__bf16)(acc[j] + bb);
  }
}

// ---------------------------------------------------------------------------
// Kernel B: flash-style attention with bf16 WMMA, f32 accumulation.
// Per workgroup: one (batch, 128-row q-tile). 8 waves; wave owns 16 q rows.
// 64-wide K tiles: per iteration 16 S-WMMAs, 16 PV-WMMAs, 2 rowsum-WMMAs.
// Row sums are produced by multiplying P against a constant all-ones B
// fragment (accL), so only the row-max needs cross-lane shuffles.
// Emits unnormalized O plus per-row running max m_q and sum l_q.
// ---------------------------------------------------------------------------
__global__ void __launch_bounds__(256) attn_kernel(
    const __bf16* __restrict__ Qb, const __bf16* __restrict__ Kb,
    const __bf16* __restrict__ Vt, float* __restrict__ Ows,
    float* __restrict__ mws, float* __restrict__ lws)
{
  __shared__ __align__(16) __bf16 ldsK[KTILE * CD];     // [k][c]   16 KB
  __shared__ __align__(16) __bf16 ldsV[CD * KTILE];     // [d][k]   16 KB
  __shared__ __align__(16) __bf16 ldsP[8][16 * KTILE];  // per-wave 16 KB

  const int tid  = threadIdx.x;
  const int wv   = tid >> 5;
  const int lane = tid & 31;
  const int n    = lane & 15;
  const int half = lane >> 4;

  const int b  = blockIdx.x >> 5;          // 32 q-tiles per batch
  const int q0 = (blockIdx.x & 31) * QTILE;
  const int qw = q0 + wv * 16;

  const unsigned int* Qd = (const unsigned int*)Qb;
  const uint4* K4 = (const uint4*)Kb;
  const uint4* V4 = (const uint4*)Vt;
  unsigned int* LK = (unsigned int*)ldsK;
  unsigned int* LV = (unsigned int*)ldsV;
  uint4* LK4 = (uint4*)ldsK;
  uint4* LV4 = (uint4*)ldsV;

  // Q A-fragments (row m = lane%16) held resident for the whole kernel.
  BFrag qf[4];
  {
    const unsigned int* qrow = Qd + ((size_t)(b * HW) + qw + n) * (CD / 2);
#pragma unroll
    for (int j = 0; j < 4; ++j) load_frag_row(qf[j], qrow, j * 16, half);
  }

  // Constant all-ones bf16 B fragment for WMMA row sums (1.0bf16 = 0x3F80).
  BFrag onesf;
#pragma unroll
  for (int i = 0; i < 8; ++i) onesf.u[i] = 0x3F803F80u;

  v8f zero = {0.f, 0.f, 0.f, 0.f, 0.f, 0.f, 0.f, 0.f};
  v8f accO[8];
#pragma unroll
  for (int j = 0; j < 8; ++j) accO[j] = zero;
  v8f accL = zero;            // running row sums (duplicated across lanes)
  float mst[8];
#pragma unroll
  for (int r = 0; r < 8; ++r) mst[r] = -1e30f;

  for (int k0 = 0; k0 < HW; k0 += KTILE) {
    // Stage K tile (contiguous 16 KB) and V tile into LDS via b128 copies.
    {
      const uint4* src = K4 + ((size_t)(b * HW) + k0) * (CD / 8);
#pragma unroll
      for (int i = 0; i < 4; ++i) LK4[tid + (i << 8)] = src[tid + (i << 8)];
#pragma unroll
      for (int i = 0; i < 4; ++i) {
        int idx = tid + (i << 8);
        int d = idx >> 3, jj = idx & 7;
        LV4[idx] = V4[((size_t)(b * CD + d)) * (HW / 8) + (k0 >> 3) + jj];
      }
    }
    __syncthreads();

    // Prefetch next iteration's K/V tiles while computing (global_prefetch_b8).
    if (k0 + KTILE < HW) {
      const char* nk = (const char*)(K4 + ((size_t)(b * HW) + k0 + KTILE) * (CD / 8));
      __builtin_prefetch(nk + tid * 64, 0, 1);
      const char* nv = (const char*)(V4 + ((size_t)(b * CD + (tid >> 1))) * (HW / 8)
                                     + ((k0 + KTILE) >> 3));
      __builtin_prefetch(nv + (tid & 1) * 64, 0, 1);
    }

    // S[16q x 64k] = Q(16x128) * K^T(128x64): four 16-col blocks, 4 dblks each.
    v8f sA[4];
#pragma unroll
    for (int e = 0; e < 4; ++e) {
      v8f s = zero;
#pragma unroll
      for (int j = 0; j < 4; ++j) {
        BFrag kb;
        load_frag_row(kb, LK + (e * 16 + n) * (CD / 2), j * 16, half);
        s = __builtin_amdgcn_wmma_f32_16x16x32_bf16(false, qf[j].v, false, kb.v,
                                                    (short)0, s, false, false);
      }
      sA[e] = s;
    }

    // Online softmax update (row r in VGPR r; lane holds rows r + 8*half).
    float scale[8];
#pragma unroll
    for (int r = 0; r < 8; ++r) {
      float v = fmaxf(fmaxf(sA[0][r], sA[1][r]), fmaxf(sA[2][r], sA[3][r]));
      v = fmaxf(v, __shfl_xor(v, 1));
      v = fmaxf(v, __shfl_xor(v, 2));
      v = fmaxf(v, __shfl_xor(v, 4));
      v = fmaxf(v, __shfl_xor(v, 8));
      float mnew = fmaxf(mst[r], v);
      scale[r] = __expf(mst[r] - mnew);
      int row = r + 8 * half;
#pragma unroll
      for (int e = 0; e < 4; ++e) {
        float p = __expf(sA[e][r] - mnew);
        ldsP[wv][row * KTILE + e * 16 + n] = (__bf16)p;   // C-layout -> LDS
      }
      mst[r] = mnew;
    }

    asm volatile("s_wait_dscnt 0" ::: "memory");  // P stores -> P reloads

    // Rescale O and row-sum accumulators by exp(m_old - m_new).
#pragma unroll
    for (int j = 0; j < 8; ++j)
#pragma unroll
      for (int r = 0; r < 8; ++r) accO[j][r] *= scale[r];
#pragma unroll
    for (int r = 0; r < 8; ++r) accL[r] *= scale[r];

    // Reload P as two A-fragments (kk 0..31 and 32..63).
    BFrag pf0, pf1;
    {
      const unsigned int* prow = (const unsigned int*)&ldsP[wv][0] + n * (KTILE / 2);
      load_frag_row(pf0, prow, 0, half);
      load_frag_row(pf1, prow, 16, half);
    }

    // Row sums: accL += P * ones  (no LDS, no shuffles).
    accL = __builtin_amdgcn_wmma_f32_16x16x32_bf16(false, pf0.v, false, onesf.v,
                                                   (short)0, accL, false, false);
    accL = __builtin_amdgcn_wmma_f32_16x16x32_bf16(false, pf1.v, false, onesf.v,
                                                   (short)0, accL, false, false);

    // O[16q x 128d] += P(16x64) * V(64x128): 16 WMMAs.
#pragma unroll
    for (int j = 0; j < 8; ++j) {
      BFrag vb;
      load_frag_row(vb, LV + (j * 16 + n) * (KTILE / 2), 0, half);
      accO[j] = __builtin_amdgcn_wmma_f32_16x16x32_bf16(false, pf0.v, false, vb.v,
                                                        (short)0, accO[j], false, false);
      load_frag_row(vb, LV + (j * 16 + n) * (KTILE / 2), 16, half);
      accO[j] = __builtin_amdgcn_wmma_f32_16x16x32_bf16(false, pf1.v, false, vb.v,
                                                        (short)0, accO[j], false, false);
    }
    __syncthreads();
  }

  // Epilogue: unnormalized O + row stats.
#pragma unroll
  for (int j = 0; j < 8; ++j)
#pragma unroll
    for (int r = 0; r < 8; ++r) {
      int q = qw + r + 8 * half;
      Ows[((size_t)(b * HW) + q) * CD + j * 16 + n] = accO[j][r];
    }
  if (n == 0) {
#pragma unroll
    for (int r = 0; r < 8; ++r) {
      int q = qw + r + 8 * half;
      mws[b * HW + q] = mst[r];
      lws[b * HW + q] = accL[r];
    }
  }
}

// ---------------------------------------------------------------------------
// Kernel C: per-batch global max M and partition Z for the global softmax.
// ---------------------------------------------------------------------------
__global__ void __launch_bounds__(256) reduce_kernel(
    const float* __restrict__ mws, const float* __restrict__ lws,
    float* __restrict__ MZ)
{
  __shared__ float red[256];
  const int b = blockIdx.x, t = threadIdx.x;
  float lm = -1e30f;
  for (int i = t; i < HW; i += 256) lm = fmaxf(lm, mws[b * HW + i]);
  red[t] = lm; __syncthreads();
  for (int s = 128; s > 0; s >>= 1) {
    if (t < s) red[t] = fmaxf(red[t], red[t + s]);
    __syncthreads();
  }
  const float M = red[0];
  __syncthreads();
  float ls = 0.f;
  for (int i = t; i < HW; i += 256)
    ls += lws[b * HW + i] * __expf(mws[b * HW + i] - M);
  red[t] = ls; __syncthreads();
  for (int s = 128; s > 0; s >>= 1) {
    if (t < s) red[t] += red[t + s];
    __syncthreads();
  }
  if (t == 0) { MZ[b * 2] = M; MZ[b * 2 + 1] = red[0]; }
}

// ---------------------------------------------------------------------------
// Kernel D: out = w_out @ (O * exp(m-M)/Z) + b_out + x1
// ---------------------------------------------------------------------------
__global__ void __launch_bounds__(256) final_kernel(
    const float* __restrict__ Ows, const float* __restrict__ mws,
    const float* __restrict__ MZ, const float* __restrict__ w_out,
    const float* __restrict__ b_out, const float* __restrict__ x1,
    float* __restrict__ out)
{
  __shared__ float os[64][129];
  const int t  = threadIdx.x;
  const int b  = blockIdx.x >> 6;
  const int q0 = (blockIdx.x & 63) << 6;
  const float M    = MZ[b * 2];
  const float Zinv = 1.0f / MZ[b * 2 + 1];

#pragma unroll
  for (int i = 0; i < 32; ++i) {
    int idx = t + (i << 8);
    int q = idx >> 7, c = idx & 127;
    float sc = __expf(mws[b * HW + q0 + q] - M) * Zinv;
    os[q][c] = Ows[((size_t)(b * HW) + q0 + q) * CD + c] * sc;
  }
  __syncthreads();

  const int co = t;  // 0..255 output channel
  float acc[64];
#pragma unroll
  for (int q = 0; q < 64; ++q) acc[q] = 0.f;
  for (int c = 0; c < CD; ++c) {
    float wvv = w_out[co * CD + c];
#pragma unroll
    for (int q = 0; q < 64; ++q) acc[q] += wvv * os[q][c];
  }
  const float bb = b_out[co];
  const size_t base = ((size_t)(b * CIN) + co) * HW + q0;
#pragma unroll
  for (int q = 0; q < 64; ++q)
    out[base + q] = acc[q] + bb + x1[base + q];
}

// ---------------------------------------------------------------------------
extern "C" void kernel_launch(void* const* d_in, const int* in_sizes, int n_in,
                              void* d_out, int out_size, void* d_ws, size_t ws_size,
                              hipStream_t stream)
{
  const float* x1    = (const float*)d_in[0];
  const float* x2    = (const float*)d_in[1];
  const float* w_th  = (const float*)d_in[2];
  const float* b_th  = (const float*)d_in[3];
  const float* w_pi  = (const float*)d_in[4];
  const float* b_pi  = (const float*)d_in[5];
  const float* w_g   = (const float*)d_in[6];
  const float* b_g   = (const float*)d_in[7];
  const float* w_out = (const float*)d_in[8];
  const float* b_out = (const float*)d_in[9];
  float* out = (float*)d_out;

  char* ws = (char*)d_ws;
  size_t off = 0;
  __bf16* Qb = (__bf16*)(ws + off); off += (size_t)NB * HW * CD * 2;  // 8 MB
  __bf16* Kb = (__bf16*)(ws + off); off += (size_t)NB * HW * CD * 2;  // 8 MB
  __bf16* Vt = (__bf16*)(ws + off); off += (size_t)NB * HW * CD * 2;  // 8 MB
  float*  Ows= (float*)(ws + off);  off += (size_t)NB * HW * CD * 4;  // 16 MB
  float*  mws= (float*)(ws + off);  off += (size_t)NB * HW * 4;
  float*  lws= (float*)(ws + off);  off += (size_t)NB * HW * 4;
  float*  MZ = (float*)(ws + off);  off += 64;

  dim3 blk(256);
  proj_kernel<<<dim3(NB * 64), blk, 0, stream>>>(x1, w_th, b_th, Qb, 1);
  proj_kernel<<<dim3(NB * 64), blk, 0, stream>>>(x2, w_pi, b_pi, Kb, 1);
  proj_kernel<<<dim3(NB * 64), blk, 0, stream>>>(x2, w_g,  b_g,  Vt, 0);
  attn_kernel<<<dim3(NB * (HW / QTILE)), blk, 0, stream>>>(Qb, Kb, Vt, Ows, mws, lws);
  reduce_kernel<<<dim3(NB), blk, 0, stream>>>(mws, lws, MZ);
  final_kernel<<<dim3(NB * 64), blk, 0, stream>>>(Ows, mws, MZ, w_out, b_out, x1, out);
}
